// ContrastiveSWM_24713241822151
// MI455X (gfx1250) — matrix operand order repaired and checked by reading, not compiled, not run on hardware
//
#include <hip/hip_runtime.h>
#include <hip/hip_bf16.h>

typedef __attribute__((ext_vector_type(16))) _Float16 v16h;
typedef __attribute__((ext_vector_type(8)))  _Float16 v8h;
typedef __attribute__((ext_vector_type(8)))  float    v8f;

#define TILE_M 128
#define TILE_N 128
#define TILE_K 32
#define LDSW   40   // padded LDS row stride in halves (80B) -> 16B-aligned frag loads

// Async DMA one 16-byte packet global -> LDS (CDNA5, tracked by ASYNCcnt).
__device__ __forceinline__ void async_copy_b128(unsigned lds_off, const _Float16* g)
{
    asm volatile("global_load_async_to_lds_b128 %0, %1, off"
                 :: "v"(lds_off), "v"((unsigned long long)(size_t)g)
                 : "memory");
}

// ---------------------------------------------------------------------------
// Tiled f16 WMMA GEMM: C[M,N](f16) = epilogue(A[M,Kp] @ Bt[N,Kp]^T + bias)
//   A  : f16, row-major (M x Kp), lda in halves
//   Bt : f16, N-MAJOR weights (N x Kp), ldb = Kp  (pre-transposed at weight prep)
// M%128==0, N%128==0, Kp%32==0. 256 thr = 8 waves (2M x 4N); wave = 64x32 via 4x2 WMMA.
// Double-buffered LDS tiles filled by global_load_async_to_lds_b128 (software pipeline).
// ---------------------------------------------------------------------------
__global__ __launch_bounds__(256) void gemm_f16_wmma(
    const _Float16* __restrict__ A, int lda,
    const _Float16* __restrict__ Bt, int ldb,
    _Float16* __restrict__ C, int ldc,
    const float* __restrict__ bias,
    int M, int N, int Kp, int relu)
{
    __shared__ __align__(16) _Float16 sA[2][TILE_M * LDSW];   // [buf][row][k]
    __shared__ __align__(16) _Float16 sB[2][TILE_N * LDSW];   // [buf][n][k]

    const int tid   = threadIdx.x;
    const int wave  = tid >> 5;
    const int lane  = tid & 31;
    const int waveM = wave >> 2;          // 0..1
    const int waveN = wave & 3;           // 0..3
    const int hi    = lane >> 4;          // 0/1
    const int ml    = lane & 15;

    const int blockM = blockIdx.y * TILE_M;
    const int blockN = blockIdx.x * TILE_N;

    v8f zero = {0.f,0.f,0.f,0.f,0.f,0.f,0.f,0.f};
    v8f acc[4][2];
#pragma unroll
    for (int i = 0; i < 4; ++i)
#pragma unroll
        for (int j = 0; j < 2; ++j) acc[i][j] = zero;

    // tile staging mapping: thread t covers (row = t>>1, 16 halves at (t&1)*16)
    const int srow = tid >> 1;            // 0..127
    const int scol = (tid & 1) * 16;      // 0 / 16

    const _Float16* gA = A  + (size_t)(blockM + srow) * lda + scol;
    const _Float16* gB = Bt + (size_t)(blockN + srow) * ldb + scol;
    unsigned ldsA[2], ldsB[2];
    ldsA[0] = (unsigned)(size_t)&sA[0][srow * LDSW + scol];
    ldsA[1] = (unsigned)(size_t)&sA[1][srow * LDSW + scol];
    ldsB[0] = (unsigned)(size_t)&sB[0][srow * LDSW + scol];
    ldsB[1] = (unsigned)(size_t)&sB[1][srow * LDSW + scol];

    const int kTiles = Kp / TILE_K;

    // prologue: async-fill buffer 0 with tile 0 (4 x b128 per thread)
    async_copy_b128(ldsA[0],      gA);
    async_copy_b128(ldsA[0] + 16, gA + 8);
    async_copy_b128(ldsB[0],      gB);
    async_copy_b128(ldsB[0] + 16, gB + 8);

    for (int kt = 0; kt < kTiles; ++kt) {
        const int cur = kt & 1;

        if (kt + 1 < kTiles) {
            // issue next tile into the other buffer, then wait for current tile's
            // 4 copies (async ops complete in order per wave => <=4 outstanding)
            const int koff = (kt + 1) * TILE_K;
            async_copy_b128(ldsA[cur ^ 1],      gA + koff);
            async_copy_b128(ldsA[cur ^ 1] + 16, gA + koff + 8);
            async_copy_b128(ldsB[cur ^ 1],      gB + koff);
            async_copy_b128(ldsB[cur ^ 1] + 16, gB + koff + 8);
            asm volatile("s_wait_asynccnt 0x4" ::: "memory");
        } else {
            asm volatile("s_wait_asynccnt 0x0" ::: "memory");
        }
        __syncthreads();    // all waves' async copies for 'cur' complete

        // --- fragment loads per documented wave32 VGPR layouts ---
        v16h afrag[4];
#pragma unroll
        for (int i = 0; i < 4; ++i) {
            const _Float16* base = &sA[cur][(waveM * 64 + i * 16 + ml) * LDSW];
            v8h lo = *(const v8h*)(base + hi * 8);        // K = hi*8 .. hi*8+7
            v8h h2 = *(const v8h*)(base + hi * 8 + 16);   // K = hi*8+16 .. +23
            afrag[i] = __builtin_shufflevector(lo, h2, 0,1,2,3,4,5,6,7,8,9,10,11,12,13,14,15);
        }
        v16h bfrag[2];
#pragma unroll
        for (int j = 0; j < 2; ++j) {
            const _Float16* base = &sB[cur][(waveN * 32 + j * 16 + ml) * LDSW];
            v8h lo = *(const v8h*)(base + hi * 8);        // K = hi*8 .. hi*8+7
            v8h h2 = *(const v8h*)(base + hi * 8 + 16);   // K = hi*8+16 .. +23
            bfrag[j] = __builtin_shufflevector(lo, h2, 0,1,2,3,4,5,6,7,8,9,10,11,12,13,14,15);
        }

#pragma unroll
        for (int i = 0; i < 4; ++i)
#pragma unroll
            for (int j = 0; j < 2; ++j)
                acc[i][j] = __builtin_amdgcn_wmma_f32_16x16x32_f16(
                    false, afrag[i], false, bfrag[j], (short)0, acc[i][j], false, false);

        __syncthreads();    // reads of 'cur' done before next iter overwrites it
    }

    // --- epilogue: D layout (lane<16: M=r, N=lane ; lane>=16: M=8+r, N=lane-16) ---
#pragma unroll
    for (int i = 0; i < 4; ++i) {
#pragma unroll
        for (int j = 0; j < 2; ++j) {
            const int col = blockN + waveN * 32 + j * 16 + ml;
            const float bv = bias[col];
#pragma unroll
            for (int r = 0; r < 8; ++r) {
                const int row = blockM + waveM * 64 + i * 16 + hi * 8 + r;
                float v = acc[i][j][r] + bv;
                if (relu) v = v > 0.f ? v : 0.f;
                C[(size_t)row * ldc + col] = (_Float16)v;
            }
        }
    }
}

// ---------------------------------------------------------------------------
// Elementwise / reduction helpers
// ---------------------------------------------------------------------------
__global__ void fill_zero_f16(_Float16* p, int n)
{
    int i = blockIdx.x * blockDim.x + threadIdx.x;
    if (i < n) p[i] = (_Float16)0.f;
}

// obs (B,3,50,50) f32 -> im2col A (102400 x 320) f16 (non-overlapping patches)
__global__ void im2col_conv1(const float* __restrict__ obs, _Float16* __restrict__ A)
{
    int idx = blockIdx.x * blockDim.x + threadIdx.x;
    if (idx >= 102400 * 320) return;
    int k = idx % 320, m = idx / 320;
    _Float16 v = (_Float16)0.f;
    if (k < 300) {
        int c = k / 100, rr = (k % 100) / 10, cc = k % 10;
        int b = m / 25, p = m % 25, oy = p / 5, ox = p % 5;
        v = (_Float16)obs[(((size_t)b * 3 + c) * 50 + oy * 10 + rr) * 50 + ox * 10 + cc];
    }
    A[idx] = v;
}

// conv1_w (512,3,10,10) f32 -> N-major (512 x 320) f16, zero-padded k >= 300
__global__ void conv1_weight(const float* __restrict__ w, _Float16* __restrict__ dst)
{
    int idx = blockIdx.x * blockDim.x + threadIdx.x;
    if (idx >= 512 * 320) return;
    int k = idx % 320, n = idx / 320;
    dst[idx] = (k < 300) ? (_Float16)w[(size_t)n * 300 + k] : (_Float16)0.f;
}

// generic (Ksrc x N) f32 -> N-MAJOR (N x Kpad) f16, zero-padded K
__global__ void weight_pad(const float* __restrict__ src, _Float16* __restrict__ dst,
                           int Ksrc, int Kpad, int N)
{
    int idx = blockIdx.x * blockDim.x + threadIdx.x;
    if (idx >= N * Kpad) return;
    int k = idx % Kpad, n = idx / Kpad;
    dst[idx] = (k < Ksrc) ? (_Float16)src[(size_t)k * N + n] : (_Float16)0.f;
}

// per-channel batch-norm stats over M=102400 samples, 512 channels
__global__ __launch_bounds__(256) void bn_stats(const _Float16* __restrict__ H,
                                                float* __restrict__ mu,
                                                float* __restrict__ rinv, int M)
{
    int ch = blockIdx.x;
    float s = 0.f, s2 = 0.f;
    for (int m = threadIdx.x; m < M; m += 256) {
        float v = (float)H[(size_t)m * 512 + ch];
        s += v; s2 += v * v;
    }
    __shared__ float sh[2][256];
    sh[0][threadIdx.x] = s; sh[1][threadIdx.x] = s2;
    __syncthreads();
    for (int off = 128; off > 0; off >>= 1) {
        if ((int)threadIdx.x < off) {
            sh[0][threadIdx.x] += sh[0][threadIdx.x + off];
            sh[1][threadIdx.x] += sh[1][threadIdx.x + off];
        }
        __syncthreads();
    }
    if (threadIdx.x == 0) {
        float mean = sh[0][0] / (float)M;
        float var  = sh[1][0] / (float)M - mean * mean;
        mu[ch] = mean;
        rinv[ch] = rsqrtf(var + 1e-5f);
    }
}

// fused: bn -> relu -> 1x1 conv (512->5) -> sigmoid -> pack into enc input (20480 x 32)
__global__ void bn_conv2_sigmoid(const _Float16* __restrict__ H,
                                 const float* __restrict__ mu, const float* __restrict__ rinv,
                                 const float* __restrict__ gamma, const float* __restrict__ beta,
                                 const float* __restrict__ w2, const float* __restrict__ b2,
                                 _Float16* __restrict__ Xenc)
{
    int idx = blockIdx.x * blockDim.x + threadIdx.x;    // m*5 + ko
    if (idx >= 102400 * 5) return;
    int ko = idx % 5, m = idx / 5;
    const _Float16* h = H + (size_t)m * 512;
    const float* w = w2 + ko * 512;
    float s = 0.f;
    for (int n = 0; n < 512; ++n) {
        float v = ((float)h[n] - mu[n]) * rinv[n] * gamma[n] + beta[n];
        v = v > 0.f ? v : 0.f;
        s += v * w[n];
    }
    s += b2[ko];
    float sig = 1.f / (1.f + __expf(-s));
    int b = m / 25, p = m % 25;
    Xenc[((size_t)(b * 5 + ko)) * 32 + p] = (_Float16)sig;
}

// row-wise LayerNorm(512) + relu, in place, one wave per row
__global__ __launch_bounds__(256) void ln_relu_512(_Float16* __restrict__ X,
                                                   const float* __restrict__ g,
                                                   const float* __restrict__ b, int M)
{
    int wave = threadIdx.x >> 5, lane = threadIdx.x & 31;
    int row = blockIdx.x * 8 + wave;
    if (row >= M) return;
    _Float16* p = X + (size_t)row * 512;
    float x[16], s = 0.f;
#pragma unroll
    for (int i = 0; i < 16; ++i) { x[i] = (float)p[lane + i * 32]; s += x[i]; }
    for (int off = 16; off > 0; off >>= 1) s += __shfl_xor(s, off, 32);
    float mean = s * (1.f / 512.f);
    float vs = 0.f;
#pragma unroll
    for (int i = 0; i < 16; ++i) { float d = x[i] - mean; vs += d * d; }
    for (int off = 16; off > 0; off >>= 1) vs += __shfl_xor(vs, off, 32);
    float rstd = rsqrtf(vs * (1.f / 512.f) + 1e-5f);
#pragma unroll
    for (int i = 0; i < 16; ++i) {
        int c = lane + i * 32;
        float v = (x[i] - mean) * rstd * g[c] + b[c];
        p[c] = (_Float16)(v > 0.f ? v : 0.f);
    }
}

// edge inputs: Xe[(b*20+e), 0:128]=state[b,row[e]], [128:256]=state[b,col[e]]
__global__ void build_edges(const _Float16* __restrict__ state, _Float16* __restrict__ Xe)
{
    int idx = blockIdx.x * blockDim.x + threadIdx.x;    // 81920 * 256
    if (idx >= 81920 * 256) return;
    int c = idx & 255, be = idx >> 8;
    int b = be / 20, e = be % 20;
    int i = e >> 2, q = e & 3;
    int j = (q < i) ? q : q + 1;
    int node = (c < 128) ? i : j;
    Xe[idx] = state[((size_t)(b * 5 + node)) * 128 + (c & 127)];
}

// node inputs: [state(128) | act_onehot(4) | agg(512) | pad(28)] = 672 cols
// agg[b,k,:] = sum of the 4 edge rows with row[e]==k (e = 4k..4k+3) -> no atomics
__global__ void build_node(const _Float16* __restrict__ state,
                           const int* __restrict__ action,
                           const _Float16* __restrict__ Eout,
                           _Float16* __restrict__ Xn)
{
    int idx = blockIdx.x * blockDim.x + threadIdx.x;    // 20480 * 672
    if (idx >= 20480 * 672) return;
    int c = idx % 672, r = idx / 672;
    int b = r / 5, k = r % 5;
    _Float16 v = (_Float16)0.f;
    if (c < 128) {
        v = state[(size_t)r * 128 + c];
    } else if (c < 132) {
        int a = c - 128;
        v = (_Float16)((action[b] == k * 4 + a) ? 1.f : 0.f);
    } else if (c < 644) {
        int n = c - 132;
        float s = 0.f;
#pragma unroll
        for (int q = 0; q < 4; ++q)
            s += (float)Eout[((size_t)(b * 20 + k * 4 + q)) * 512 + n];
        v = (_Float16)s;
    }
    Xn[(size_t)r * 672 + c] = v;
}

__global__ void final_add(const _Float16* __restrict__ state,
                          const _Float16* __restrict__ delta,
                          float* __restrict__ out, int n)
{
    int i = blockIdx.x * blockDim.x + threadIdx.x;
    if (i < n) out[i] = (float)state[i] + (float)delta[i];
}

// ---------------------------------------------------------------------------
// Host launch
// ---------------------------------------------------------------------------
static inline dim3 grid1d(long long n) { return dim3((unsigned)((n + 255) / 256)); }

extern "C" void kernel_launch(void* const* d_in, const int* in_sizes, int n_in,
                              void* d_out, int out_size, void* d_ws, size_t ws_size,
                              hipStream_t stream)
{
    const float* obs      = (const float*)d_in[0];
    const int*   action   = (const int*)  d_in[1];
    const float* conv1_w  = (const float*)d_in[2];
    const float* conv1_b  = (const float*)d_in[3];
    const float* bn1_g    = (const float*)d_in[4];
    const float* bn1_b    = (const float*)d_in[5];
    const float* conv2_w  = (const float*)d_in[6];
    const float* conv2_b  = (const float*)d_in[7];
    const float* enc_w1   = (const float*)d_in[8];
    const float* enc_b1   = (const float*)d_in[9];
    const float* enc_w2   = (const float*)d_in[10];
    const float* enc_b2   = (const float*)d_in[11];
    const float* enc_lng  = (const float*)d_in[12];
    const float* enc_lnb  = (const float*)d_in[13];
    const float* enc_w3   = (const float*)d_in[14];
    const float* enc_b3   = (const float*)d_in[15];
    const float* edge_w1  = (const float*)d_in[16];
    const float* edge_b1  = (const float*)d_in[17];
    const float* edge_w2  = (const float*)d_in[18];
    const float* edge_b2  = (const float*)d_in[19];
    const float* edge_lng = (const float*)d_in[20];
    const float* edge_lnb = (const float*)d_in[21];
    const float* edge_w3  = (const float*)d_in[22];
    const float* edge_b3  = (const float*)d_in[23];
    const float* node_w1  = (const float*)d_in[24];
    const float* node_b1  = (const float*)d_in[25];
    const float* node_w2  = (const float*)d_in[26];
    const float* node_b2  = (const float*)d_in[27];
    const float* node_lng = (const float*)d_in[28];
    const float* node_lnb = (const float*)d_in[29];
    const float* node_w3  = (const float*)d_in[30];
    const float* node_b3  = (const float*)d_in[31];
    float* out = (float*)d_out;

    _Float16* ws = (_Float16*)d_ws;

    // ---- workspace layout (halves). Conv-phase region is aliased with edge-phase. ----
    const size_t REGION0 = 104857600;                 // max(conv: 85.2M, edge: 104.9M)
    _Float16* convA = ws + 0;                         // 102400*320 = 32,768,000
    _Float16* convH = ws + 32768000;                  // 102400*512 = 52,428,800
    _Float16* Xe    = ws + 0;                         // 81920*256  = 20,971,520
    _Float16* Eb1   = ws + 20971520;                  // 81920*512  = 41,943,040
    _Float16* Eb2   = ws + 62914560;                  // 81920*512  = 41,943,040

    size_t o = REGION0;
    _Float16* Wc1   = ws + o; o += 320 * 512;
    _Float16* We1   = ws + o; o += 32  * 512;
    _Float16* We2   = ws + o; o += 512 * 512;
    _Float16* We3   = ws + o; o += 512 * 128;
    _Float16* Wg1   = ws + o; o += 256 * 512;
    _Float16* Wg2   = ws + o; o += 512 * 512;
    _Float16* Wg3   = ws + o; o += 512 * 512;
    _Float16* Wn1   = ws + o; o += 672 * 512;
    _Float16* Wn2   = ws + o; o += 512 * 512;
    _Float16* Wn3   = ws + o; o += 512 * 128;
    _Float16* Xenc  = ws + o; o += 20480 * 32;
    _Float16* T1    = ws + o; o += 20480 * 512;
    _Float16* T2    = ws + o; o += 20480 * 512;
    _Float16* St    = ws + o; o += 20480 * 128;
    _Float16* Xn    = ws + o; o += 20480 * 672;
    _Float16* Dl    = ws + o; o += 20480 * 128;
    float*    mu    = (float*)(ws + o);               // 512 f32
    float*    rinv  = mu + 512;                       // 512 f32

    const dim3 blk(256);

    // ---- weight conversion to f16, N-MAJOR (N x Kpad), K padded to mult of 32 ----
    conv1_weight<<<grid1d(512 * 320), blk, 0, stream>>>(conv1_w, Wc1);
    weight_pad<<<grid1d(512 * 32 ), blk, 0, stream>>>(enc_w1, We1, 25, 32, 512);
    weight_pad<<<grid1d(512 * 512), blk, 0, stream>>>(enc_w2, We2, 512, 512, 512);
    weight_pad<<<grid1d(128 * 512), blk, 0, stream>>>(enc_w3, We3, 512, 512, 128);
    weight_pad<<<grid1d(512 * 256), blk, 0, stream>>>(edge_w1, Wg1, 256, 256, 512);
    weight_pad<<<grid1d(512 * 512), blk, 0, stream>>>(edge_w2, Wg2, 512, 512, 512);
    weight_pad<<<grid1d(512 * 512), blk, 0, stream>>>(edge_w3, Wg3, 512, 512, 512);
    weight_pad<<<grid1d(512 * 672), blk, 0, stream>>>(node_w1, Wn1, 644, 672, 512);
    weight_pad<<<grid1d(512 * 512), blk, 0, stream>>>(node_w2, Wn2, 512, 512, 512);
    weight_pad<<<grid1d(128 * 512), blk, 0, stream>>>(node_w3, Wn3, 512, 512, 128);

    // ---- conv1 (implicit GEMM on non-overlapping patches) + BN + relu + conv2 + sigmoid ----
    im2col_conv1<<<grid1d(102400LL * 320), blk, 0, stream>>>(obs, convA);
    gemm_f16_wmma<<<dim3(4, 800), blk, 0, stream>>>(convA, 320, Wc1, 320, convH, 512,
                                                    conv1_b, 102400, 512, 320, 0);
    bn_stats<<<dim3(512), blk, 0, stream>>>(convH, mu, rinv, 102400);
    fill_zero_f16<<<grid1d(20480 * 32), blk, 0, stream>>>(Xenc, 20480 * 32);
    bn_conv2_sigmoid<<<grid1d(102400 * 5), blk, 0, stream>>>(convH, mu, rinv, bn1_g, bn1_b,
                                                             conv2_w, conv2_b, Xenc);

    // ---- encoder MLP3: 32 -> 512 -> 512(LN) -> 128 ----
    gemm_f16_wmma<<<dim3(4, 160), blk, 0, stream>>>(Xenc, 32, We1, 32, T1, 512, enc_b1, 20480, 512, 32, 1);
    gemm_f16_wmma<<<dim3(4, 160), blk, 0, stream>>>(T1, 512, We2, 512, T2, 512, enc_b2, 20480, 512, 512, 0);
    ln_relu_512<<<dim3(20480 / 8), blk, 0, stream>>>(T2, enc_lng, enc_lnb, 20480);
    gemm_f16_wmma<<<dim3(1, 160), blk, 0, stream>>>(T2, 512, We3, 512, St, 128, enc_b3, 20480, 128, 512, 0);

    // ---- edge MLP3: 256 -> 512 -> 512(LN) -> 512  (conv buffers are dead now) ----
    build_edges<<<grid1d(81920LL * 256), blk, 0, stream>>>(St, Xe);
    gemm_f16_wmma<<<dim3(4, 640), blk, 0, stream>>>(Xe, 256, Wg1, 256, Eb1, 512, edge_b1, 81920, 512, 256, 1);
    gemm_f16_wmma<<<dim3(4, 640), blk, 0, stream>>>(Eb1, 512, Wg2, 512, Eb2, 512, edge_b2, 81920, 512, 512, 0);
    ln_relu_512<<<dim3(81920 / 8), blk, 0, stream>>>(Eb2, edge_lng, edge_lnb, 81920);
    gemm_f16_wmma<<<dim3(4, 640), blk, 0, stream>>>(Eb2, 512, Wg3, 512, Eb1, 512, edge_b3, 81920, 512, 512, 0);

    // ---- node MLP3: 672 -> 512 -> 512(LN) -> 128, then out = state + delta ----
    build_node<<<grid1d(20480LL * 672), blk, 0, stream>>>(St, action, Eb1, Xn);
    gemm_f16_wmma<<<dim3(4, 160), blk, 0, stream>>>(Xn, 672, Wn1, 672, T1, 512, node_b1, 20480, 512, 672, 1);
    gemm_f16_wmma<<<dim3(4, 160), blk, 0, stream>>>(T1, 512, Wn2, 512, T2, 512, node_b2, 20480, 512, 512, 0);
    ln_relu_512<<<dim3(20480 / 8), blk, 0, stream>>>(T2, node_lng, node_lnb, 20480);
    gemm_f16_wmma<<<dim3(1, 160), blk, 0, stream>>>(T2, 512, Wn3, 512, Dl, 128, node_b3, 20480, 128, 512, 0);

    final_add<<<grid1d(20480 * 128), blk, 0, stream>>>(St, Dl, out, 20480 * 128);
}
